// Attention_18700287607351
// MI455X (gfx1250) — matrix operand order repaired and checked by reading, not compiled, not run on hardware
//
#include <hip/hip_runtime.h>
#include <hip/hip_bf16.h>

#define B_    2
#define T_    2048
#define D_    2048
#define H_    32
#define HKV_  8
#define HD_   64
#define REP_  (H_ / HKV_)
#define SCALE_ 0.125f      /* HD^-0.5 */
#define EPS_   1e-6f
#define THETA_ 3.0f

typedef __attribute__((ext_vector_type(16))) __bf16 v16bf;
typedef __attribute__((ext_vector_type(8)))  __bf16 v8bf;
typedef __attribute__((ext_vector_type(8)))  float  v8f;

static __device__ __forceinline__ v8f wmma_bf16(v16bf a, v16bf b, v8f c) {
  // D = A(16x32 bf16) x B(32x16 bf16) + C(16x16 f32)
  return __builtin_amdgcn_wmma_f32_16x16x32_bf16(false, a, false, b, (short)0, c,
                                                 false, false);
}

// ---- A fragment (16x32 bf16) from row-major bf16 src (16B-aligned rows).
// ISA layout: M = lane&15; lane half0 holds K 0-7 & 16-23, half1 K 8-15 & 24-31.
static __device__ __forceinline__ v16bf load_a(const __bf16* p, int lda, int lane) {
  const int m  = lane & 15;
  const int kb = (lane >> 4) ? 8 : 0;
  const v8bf lo = *(const v8bf*)(p + (size_t)m * lda + kb);        // b128
  const v8bf hh = *(const v8bf*)(p + (size_t)m * lda + 16 + kb);   // b128
  v16bf r;
#pragma unroll
  for (int j = 0; j < 8; ++j) { r[j] = lo[j]; r[8 + j] = hh[j]; }
  return r;
}

// ---- B fragment (32x16) where B[k][n] = p[n*lda + k] (contiguous in k).
// ISA layout: N = lane&15; half0 K 0-7 & 16-23, half1 K 8-15 & 24-31.
static __device__ __forceinline__ v16bf load_bt(const __bf16* p, int lda, int lane) {
  const int n  = lane & 15;
  const int kb = (lane >> 4) ? 8 : 0;
  const v8bf lo = *(const v8bf*)(p + (size_t)n * lda + kb);        // b128
  const v8bf hh = *(const v8bf*)(p + (size_t)n * lda + 16 + kb);   // b128
  v16bf r;
#pragma unroll
  for (int j = 0; j < 8; ++j) { r[j] = lo[j]; r[8 + j] = hh[j]; }
  return r;
}

// ============ one-shot fp32 -> bf16 conversion (x and weights) =================
__global__ __launch_bounds__(256) void cvtbf_kernel(const float* src, __bf16* dst, int n) {
  const int i = (blockIdx.x * blockDim.x + threadIdx.x) * 4;
  if (i >= n) return;
  const float4 v = *(const float4*)(src + i);
  dst[i]     = (__bf16)v.x;
  dst[i + 1] = (__bf16)v.y;
  dst[i + 2] = (__bf16)v.z;
  dst[i + 3] = (__bf16)v.w;
}

// ============ GEMM: C[M x N] f32 = A[M x K] @ W[N x K]^T  (all-bf16 operands) ===
// One wave computes a 64x64 tile (4x4 subtiles) -> 4x A/B register reuse.
// Consecutive waves in a block share row0, so A fragments hit the WGP cache;
// the 192MB global L2 keeps A and W resident across the grid.
__global__ __launch_bounds__(128) void gemm_bt_kernel(const __bf16* A, const __bf16* W,
                                                      float* C, int M, int N, int K) {
  const int lane   = threadIdx.x & 31;
  const int wid    = blockIdx.x * (blockDim.x >> 5) + (threadIdx.x >> 5);
  const int tilesN = N >> 6;
  const int tilesM = M >> 6;
  if (wid >= tilesM * tilesN) return;
  const int row0 = (wid / tilesN) << 6;
  const int col0 = (wid % tilesN) << 6;

  v8f acc[4][4];
#pragma unroll
  for (int i = 0; i < 4; ++i)
#pragma unroll
    for (int j = 0; j < 4; ++j) { v8f z = {}; acc[i][j] = z; }

  for (int k = 0; k < K; k += 32) {
    v16bf a[4], b[4];
#pragma unroll
    for (int i = 0; i < 4; ++i)
      a[i] = load_a(A + (size_t)(row0 + i * 16) * K + k, K, lane);
#pragma unroll
    for (int j = 0; j < 4; ++j)
      b[j] = load_bt(W + (size_t)(col0 + j * 16) * K + k, K, lane);
    if (k + 32 < K)  // pull next K-slab of W toward the caches (global_prefetch_b8)
      __builtin_prefetch(&W[(size_t)(col0 + (lane & 15) * 4) * K + k + 32], 0, 3);
#pragma unroll
    for (int i = 0; i < 4; ++i)
#pragma unroll
      for (int j = 0; j < 4; ++j) acc[i][j] = wmma_bf16(a[i], b[j], acc[i][j]);
  }

  const int hi = lane >> 4;
  const int nl = lane & 15;
#pragma unroll
  for (int i = 0; i < 4; ++i)
#pragma unroll
    for (int j = 0; j < 4; ++j)
#pragma unroll
      for (int r = 0; r < 8; ++r)
        C[(size_t)(row0 + i * 16 + r + 8 * hi) * N + col0 + j * 16 + nl] = acc[i][j][r];
}

// ============ per-head RMSNorm + RoPE, fp32 -> bf16 (head-major layout) =========
__global__ __launch_bounds__(128) void rmsrope_kernel(const float* src, const float* w,
                                                      __bf16* dst, int NH, float outScale) {
  const int lane = threadIdx.x & 31;
  const int wid  = blockIdx.x * (blockDim.x >> 5) + (threadIdx.x >> 5);
  if (wid >= B_ * T_ * NH) return;
  const int h = wid % NH;
  const int t = (wid / NH) % T_;
  const int b = wid / (NH * T_);

  const float* p = src + ((size_t)(b * T_ + t) * NH + h) * HD_;
  float x1 = p[2 * lane], x2 = p[2 * lane + 1];
  float ss = x1 * x1 + x2 * x2;
#pragma unroll
  for (int m = 16; m >= 1; m >>= 1) ss += __shfl_xor(ss, m, 32);
  const float rn = rsqrtf(ss * (1.0f / HD_) + EPS_);
  const float y1 = x1 * rn * w[2 * lane];
  const float y2 = x2 * rn * w[2 * lane + 1];

  const float invf = __powf(THETA_, -(float)(2 * lane) * (1.0f / HD_));
  float s, c;
  __sincosf((float)t * invf, &s, &c);
  __bf16* q = dst + ((size_t)(b * NH + h) * T_ + t) * HD_;
  q[2 * lane]     = (__bf16)((y1 * c - y2 * s) * outScale);
  q[2 * lane + 1] = (__bf16)((y1 * s + y2 * c) * outScale);
}

// ============ V transpose: (B,T,HKV,HD) f32 -> (B,HKV,HD,T) bf16 ===============
__global__ void vtrans_kernel(const float* src, __bf16* dst) {
  size_t i = (size_t)blockIdx.x * blockDim.x + threadIdx.x;
  if (i >= (size_t)B_ * T_ * HKV_ * HD_) return;
  const int d = i % HD_;
  const int h = (i / HD_) % HKV_;
  const int t = (i / (HD_ * HKV_)) % T_;
  const int b = i / ((size_t)HD_ * HKV_ * T_);
  dst[((size_t)(b * HKV_ + h) * HD_ + d) * T_ + t] = (__bf16)src[i];
}

// ============ Flash attention (causal, GQA) ====================================
// One 32-key step of online softmax. MASKED is instantiated only for the
// diagonal block(s); interior blocks run with zero compare/cndmask overhead.
template <bool MASKED>
static __device__ __forceinline__ void attn_step(
    int key0, int q0, int lane, int hi, int nl,
    const __bf16* kbase, const __bf16* vbase,
    const v16bf& qa0, const v16bf& qa1,
    v8f o[4], float mrow[8], float lrow[8],
    __bf16 (*pb)[32]) {
  const __bf16* kp = kbase + (size_t)key0 * HD_;
  v8f s0 = {}, s1 = {};
  s0 = wmma_bf16(qa0, load_bt(kp, HD_, lane), s0);
  s0 = wmma_bf16(qa1, load_bt(kp + 32, HD_, lane), s0);
  s1 = wmma_bf16(qa0, load_bt(kp + (size_t)16 * HD_, HD_, lane), s1);
  s1 = wmma_bf16(qa1, load_bt(kp + (size_t)16 * HD_ + 32, HD_, lane), s1);

#pragma unroll
  for (int r = 0; r < 8; ++r) {
    const int row = r + 8 * hi;  // C/D layout: vgpr r -> row r (+8 in hi half)
    float e0 = s0[r];
    float e1 = s1[r];
    if (MASKED) {
      const int qi = q0 + row;
      e0 = (key0 + nl      <= qi) ? e0 : -1e9f;
      e1 = (key0 + 16 + nl <= qi) ? e1 : -1e9f;
    }
    float mx = fmaxf(e0, e1);
#pragma unroll
    for (int msk = 8; msk >= 1; msk >>= 1) mx = fmaxf(mx, __shfl_xor(mx, msk, 32));
    const float mnew = fmaxf(mrow[r], mx);
    const float corr = __expf(mrow[r] - mnew);
    const float p0 = __expf(e0 - mnew);
    const float p1 = __expf(e1 - mnew);
    float ps = p0 + p1;
#pragma unroll
    for (int msk = 8; msk >= 1; msk >>= 1) ps += __shfl_xor(ps, msk, 32);
    lrow[r] = lrow[r] * corr + ps;
    mrow[r] = mnew;
#pragma unroll
    for (int t = 0; t < 4; ++t) o[t][r] *= corr;  // rescale running output
    pb[row][nl]      = (__bf16)p0;  // stage P row-major in LDS (D -> A transpose)
    pb[row][16 + nl] = (__bf16)p1;
  }

  // Reload P as a 16x32 A fragment (rows now across lanes); 16B-aligned b128 loads.
  v16bf pa;
  {
    const int kbq = hi ? 8 : 0;
    const v8bf lo = *(const v8bf*)&pb[nl][kbq];
    const v8bf hh = *(const v8bf*)&pb[nl][16 + kbq];
#pragma unroll
    for (int j = 0; j < 8; ++j) { pa[j] = lo[j]; pa[8 + j] = hh[j]; }
  }

  const __bf16* vp = vbase + key0;  // B[k][n] = vT[(dcol)*T + key0 + k]
#pragma unroll
  for (int t = 0; t < 4; ++t)
    o[t] = wmma_bf16(pa, load_bt(vp + (size_t)t * 16 * T_, T_, lane), o[t]);
}

__global__ __launch_bounds__(128) void attn_kernel(const __bf16* Q, const __bf16* Kc,
                                                   const __bf16* Vt, __bf16* O) {
  __shared__ __bf16 pbuf[4][16][32];
  const int lane = threadIdx.x & 31;
  const int w    = threadIdx.x >> 5;
  const int wid  = blockIdx.x * 4 + w;
  const int qtiles = T_ / 16;
  if (wid >= B_ * H_ * qtiles) return;
  const int qt  = wid % qtiles;
  const int h   = (wid / qtiles) % H_;
  const int b   = wid / (qtiles * H_);
  const int hkv = h / REP_;
  const int q0  = qt * 16;
  const int hi  = lane >> 4;
  const int nl  = lane & 15;

  const __bf16* qp = Q + ((size_t)(b * H_ + h) * T_ + q0) * HD_;
  const v16bf qa0 = load_a(qp, HD_, lane);       // d = 0..31 (softmax scale pre-folded)
  const v16bf qa1 = load_a(qp + 32, HD_, lane);  // d = 32..63

  v8f o[4];
#pragma unroll
  for (int t = 0; t < 4; ++t) { v8f z = {}; o[t] = z; }
  float mrow[8], lrow[8];
#pragma unroll
  for (int r = 0; r < 8; ++r) { mrow[r] = -1e30f; lrow[r] = 0.0f; }

  const __bf16* kbase = Kc + (size_t)(b * HKV_ + hkv) * T_ * HD_;
  const __bf16* vbase = Vt + (size_t)(b * HKV_ + hkv) * HD_ * T_;

  const int nfull = (q0 + 1) / 32;       // blocks entirely below the diagonal
  const int nkb   = (q0 + 15) / 32 + 1;  // all blocks with key <= q0+15
  for (int kb = 0; kb < nfull; ++kb)     // fast path: no mask logic at all
    attn_step<false>(kb * 32, q0, lane, hi, nl, kbase, vbase, qa0, qa1,
                     o, mrow, lrow, pbuf[w]);
  for (int kb = nfull; kb < nkb; ++kb)   // diagonal block(s): masked epilogue
    attn_step<true>(kb * 32, q0, lane, hi, nl, kbase, vbase, qa0, qa1,
                    o, mrow, lrow, pbuf[w]);

  __bf16* op = O + ((size_t)b * T_ + q0) * (H_ * HD_) + h * HD_;
#pragma unroll
  for (int t = 0; t < 4; ++t)
#pragma unroll
    for (int r = 0; r < 8; ++r) {
      const int row = r + 8 * hi;
      op[(size_t)row * (H_ * HD_) + t * 16 + nl] = (__bf16)(o[t][r] / lrow[r]);
    }
}

// ==============================================================================
extern "C" void kernel_launch(void* const* d_in, const int* in_sizes, int n_in,
                              void* d_out, int out_size, void* d_ws, size_t ws_size,
                              hipStream_t stream) {
  const float* x   = (const float*)d_in[0];
  const float* Wq  = (const float*)d_in[1];
  const float* Wk  = (const float*)d_in[2];
  const float* Wv  = (const float*)d_in[3];
  const float* Wo  = (const float*)d_in[4];
  const float* qnw = (const float*)d_in[5];
  const float* knw = (const float*)d_in[6];
  (void)in_sizes; (void)n_in; (void)out_size; (void)ws_size;

  char* ws = (char*)d_ws;
  size_t off = 0;
  float*  qf  = (float*)(ws + off);  off += (size_t)B_ * T_ * H_ * HD_ * 4;
  float*  kf  = (float*)(ws + off);  off += (size_t)B_ * T_ * HKV_ * HD_ * 4;
  float*  vf  = (float*)(ws + off);  off += (size_t)B_ * T_ * HKV_ * HD_ * 4;
  __bf16* qb  = (__bf16*)(ws + off); off += (size_t)B_ * T_ * H_ * HD_ * 2;
  __bf16* kbb = (__bf16*)(ws + off); off += (size_t)B_ * T_ * HKV_ * HD_ * 2;
  __bf16* vt  = (__bf16*)(ws + off); off += (size_t)B_ * T_ * HKV_ * HD_ * 2;
  __bf16* ab  = (__bf16*)(ws + off); off += (size_t)B_ * T_ * H_ * HD_ * 2;
  __bf16* xb  = (__bf16*)(ws + off); off += (size_t)B_ * T_ * D_ * 2;
  __bf16* wqb = (__bf16*)(ws + off); off += (size_t)H_ * HD_ * D_ * 2;
  __bf16* wkb = (__bf16*)(ws + off); off += (size_t)HKV_ * HD_ * D_ * 2;
  __bf16* wvb = (__bf16*)(ws + off); off += (size_t)HKV_ * HD_ * D_ * 2;
  __bf16* wob = (__bf16*)(ws + off); off += (size_t)D_ * H_ * HD_ * 2;

  const int M = B_ * T_;  // 4096 token rows

  // One-shot bf16 conversion of activations + weights (everything then L2-hot).
  const int nx = B_ * T_ * D_, nq = H_ * HD_ * D_, nk = HKV_ * HD_ * D_;
  cvtbf_kernel<<<nx / 1024, 256, 0, stream>>>(x,  xb,  nx);
  cvtbf_kernel<<<nq / 1024, 256, 0, stream>>>(Wq, wqb, nq);
  cvtbf_kernel<<<nk / 1024, 256, 0, stream>>>(Wk, wkb, nk);
  cvtbf_kernel<<<nk / 1024, 256, 0, stream>>>(Wv, wvb, nk);
  cvtbf_kernel<<<nq / 1024, 256, 0, stream>>>(Wo, wob, nq);

  // QKV projections (pure bf16 WMMA, f32 accumulate/out)
  gemm_bt_kernel<<<(M / 64) * (D_ / 64) / 4, 128, 0, stream>>>(
      xb, wqb, qf, M, H_ * HD_, D_);
  gemm_bt_kernel<<<(M / 64) * ((HKV_ * HD_) / 64) / 4, 128, 0, stream>>>(
      xb, wkb, kf, M, HKV_ * HD_, D_);
  gemm_bt_kernel<<<(M / 64) * ((HKV_ * HD_) / 64) / 4, 128, 0, stream>>>(
      xb, wvb, vf, M, HKV_ * HD_, D_);

  // RMSNorm + RoPE -> bf16 (softmax scale folded into Q)
  rmsrope_kernel<<<(B_ * T_ * H_) / 4, 128, 0, stream>>>(qf, qnw, qb, H_, SCALE_);
  rmsrope_kernel<<<(B_ * T_ * HKV_) / 4, 128, 0, stream>>>(kf, knw, kbb, HKV_, 1.0f);
  vtrans_kernel<<<(B_ * T_ * HKV_ * HD_) / 256, 256, 0, stream>>>(vf, vt);

  // Causal flash attention
  attn_kernel<<<(B_ * H_ * (T_ / 16)) / 4, 128, 0, stream>>>(qb, kbb, vt, ab);

  // Output projection -> fp32 d_out
  gemm_bt_kernel<<<(M / 64) * (D_ / 64) / 4, 128, 0, stream>>>(
      ab, wob, (float*)d_out, M, D_, D_);
}
